// DQN_GNN_48009144434876
// MI455X (gfx1250) — matrix operand order repaired and checked by reading, not compiled, not run on hardware
//
#include <hip/hip_runtime.h>
#include <hip/hip_bf16.h>
#include <cstdint>

#define D_HID   1024
#define NNODES  65536
#define NEDGES  65536
#define NBATCH  64
#define ETOT    (NEDGES + NNODES)

typedef __attribute__((ext_vector_type(16))) __bf16 v16bf;
typedef __attribute__((ext_vector_type(8)))  __bf16 v8bf;
typedef __attribute__((ext_vector_type(8)))  float  v8f;

struct __align__(16) B16v { unsigned long long x, y; };

static __device__ __forceinline__ uint16_t f32_to_bf16_bits(float f) {
    uint32_t u = __float_as_uint(f);
    uint32_t r = u + 0x7FFFu + ((u >> 16) & 1u);
    return (uint16_t)(r >> 16);
}
static __device__ __forceinline__ float lrelu(float x) { return x > 0.f ? x : 0.2f * x; }

// ---------------------------------------------------------------------------
// WMMA GEMM: C[M,1024] = A_bf16[M,K] @ Wt_bf16[1024,K]^T (+ bias)
// Block tile 128x128, K-step 64 (2 WMMA K-steps / barrier interval).
// FULL=true : M % 128 == 0, async global->LDS copies, double-buffered LDS.
// FULL=false: guarded synchronous staging (used for the 64-row MLP).
// ---------------------------------------------------------------------------
#define BM 128
#define BN 128
#define BK 64
#define LDT 72   // LDS row stride in bf16 elems (144B: 16B aligned, conflict pad)

template<bool FULL>
__global__ __launch_bounds__(256)
void gemm_bf16_wmma(const uint16_t* __restrict__ A,   // [M, K] bf16 row-major
                    const uint16_t* __restrict__ Bt,  // [1024, K] bf16 (= W^T)
                    const float* __restrict__ bias,   // [1024] or nullptr
                    float* __restrict__ C,            // [M, 1024] f32
                    int M, int K)
{
    __shared__ uint16_t lA[2][BM * LDT];
    __shared__ uint16_t lB[2][BN * LDT];

    const int tid  = threadIdx.x;
    const int lane = tid & 31;
    const int wave = tid >> 5;
    const int wm   = (wave & 3) * 32;   // 4 waves along M
    const int wn   = (wave >> 2) * 64;  // 2 waves along N
    const int m0   = blockIdx.x * BM;
    const int n0   = blockIdx.y * BN;

    const int halfsel = (lane < 16) ? 0 : 1;
    const int lr      = lane & 15;

    // staging geometry: 1024 16B chunks per tile, 4 per thread
    const int srow = tid >> 3;           // rows tid/8 (+32 per it step of 256)
    const int sc8  = tid & 7;            // 16B chunk within 128B row

    v8f zero = {};
    v8f acc[2][4];
#pragma unroll
    for (int i = 0; i < 2; ++i)
#pragma unroll
        for (int j = 0; j < 4; ++j) acc[i][j] = zero;

    auto stage_async = [&](int k0, int buf) {
#pragma unroll
        for (int it = 0; it < 4; ++it) {
            int row = srow + it * 32;
            unsigned la = (unsigned)(size_t)&lA[buf][row * LDT + sc8 * 8];
            unsigned long long ga =
                (unsigned long long)(size_t)(A + (size_t)(m0 + row) * K + k0 + sc8 * 8);
            asm volatile("global_load_async_to_lds_b128 %0, %1, off"
                         :: "v"(la), "v"(ga) : "memory");
            unsigned lb = (unsigned)(size_t)&lB[buf][row * LDT + sc8 * 8];
            unsigned long long gb =
                (unsigned long long)(size_t)(Bt + (size_t)(n0 + row) * K + k0 + sc8 * 8);
            asm volatile("global_load_async_to_lds_b128 %0, %1, off"
                         :: "v"(lb), "v"(gb) : "memory");
        }
    };
    auto stage_sync = [&](int k0, int buf) {
#pragma unroll
        for (int it = 0; it < 4; ++it) {
            int row = srow + it * 32;
            B16v va = {0ull, 0ull};
            int gm = m0 + row;
            if (gm < M) va = *(const B16v*)(A + (size_t)gm * K + k0 + sc8 * 8);
            *(B16v*)(&lA[buf][row * LDT + sc8 * 8]) = va;
            B16v vb = *(const B16v*)(Bt + (size_t)(n0 + row) * K + k0 + sc8 * 8);
            *(B16v*)(&lB[buf][row * LDT + sc8 * 8]) = vb;
        }
    };
    auto compute = [&](int buf) {
        const int akb = halfsel ? 8 : 0;   // A: lanes16-31 start at K=8/K=24
        const int bkb = halfsel ? 8 : 0;   // B: lanes16-31 start at K=8/K=24
#pragma unroll
        for (int kk = 0; kk < BK; kk += 32) {
            v16bf afrag[2], bfrag[4];
#pragma unroll
            for (int tm = 0; tm < 2; ++tm) {
                const uint16_t* p = &lA[buf][(wm + tm * 16 + lr) * LDT + kk + akb];
                v8bf lo = *(const v8bf*)p;
                v8bf hi = *(const v8bf*)(p + 16);
                v16bf a;
#pragma unroll
                for (int i = 0; i < 8; ++i) { a[i] = lo[i]; a[i + 8] = hi[i]; }
                afrag[tm] = a;
            }
#pragma unroll
            for (int tn = 0; tn < 4; ++tn) {
                const uint16_t* p = &lB[buf][(wn + tn * 16 + lr) * LDT + kk + bkb];
                v8bf lo = *(const v8bf*)p;
                v8bf hi = *(const v8bf*)(p + 16);
                v16bf b;
#pragma unroll
                for (int i = 0; i < 8; ++i) { b[i] = lo[i]; b[i + 8] = hi[i]; }
                bfrag[tn] = b;
            }
#pragma unroll
            for (int tm = 0; tm < 2; ++tm)
#pragma unroll
                for (int tn = 0; tn < 4; ++tn)
                    acc[tm][tn] = __builtin_amdgcn_wmma_f32_16x16x32_bf16(
                        false, afrag[tm], false, bfrag[tn],
                        (short)0, acc[tm][tn], false, false);
        }
    };

    const int nk = K / BK;
    if (FULL) {
        stage_async(0, 0);
        asm volatile("s_wait_asynccnt 0x0" ::: "memory");
        __syncthreads();
        for (int kt = 0; kt < nk; ++kt) {
            int b = kt & 1;
            if (kt + 1 < nk) stage_async((kt + 1) * BK, 1 - b);  // overlap with WMMA
            compute(b);
            asm volatile("s_wait_asynccnt 0x0" ::: "memory");
            __syncthreads();
        }
    } else {
        for (int kt = 0; kt < nk; ++kt) {
            stage_sync(kt * BK, 0);
            __syncthreads();
            compute(0);
            __syncthreads();
        }
    }

    // epilogue: C/D layout: VGPR r -> M=r (lanes 0-15) / M=8+r (lanes 16-31)
    const int rbase = halfsel * 8;
#pragma unroll
    for (int tm = 0; tm < 2; ++tm)
#pragma unroll
        for (int tn = 0; tn < 4; ++tn) {
            int col  = n0 + wn + tn * 16 + lr;
            float bv = bias ? bias[col] : 0.0f;
#pragma unroll
            for (int r = 0; r < 8; ++r) {
                int row = m0 + wm + tm * 16 + rbase + r;
                if (FULL || row < M)
                    C[(size_t)row * D_HID + col] = acc[tm][tn][r] + bv;
            }
        }
}

// ---------------------------------------------------------------------------
// Conversions
// ---------------------------------------------------------------------------
__global__ __launch_bounds__(256)
void convert_pad_bf16(const float* __restrict__ src, uint16_t* __restrict__ dst,
                      long long rows, int K, int Kpad)
{
    long long idx = (long long)blockIdx.x * 256 + threadIdx.x;
    if (idx >= rows * (long long)Kpad) return;
    int kp = (int)(idx % Kpad);
    long long row = idx / Kpad;
    float v = (kp < K) ? src[row * K + kp] : 0.0f;
    dst[idx] = f32_to_bf16_bits(v);
}

__global__ __launch_bounds__(256)
void wt_convert(const float* __restrict__ W, uint16_t* __restrict__ Wt, int K, int Kpad)
{
    int idx = blockIdx.x * 256 + threadIdx.x;   // over 1024*Kpad
    if (idx >= D_HID * Kpad) return;
    int kp = idx % Kpad;
    int n  = idx / Kpad;
    float v = (kp < K) ? W[(size_t)kp * D_HID + n] : 0.0f;
    Wt[idx] = f32_to_bf16_bits(v);
}

__global__ __launch_bounds__(256)
void fill_i32(int* p, int v, long long n)
{ long long i = (long long)blockIdx.x * 256 + threadIdx.x; if (i < n) p[i] = v; }

// ---------------------------------------------------------------------------
// One-time CSR build over dst (edge list + self loops); topology is layer-invariant
// ---------------------------------------------------------------------------
__global__ __launch_bounds__(256)
void deg_count(const int* __restrict__ ei, int* __restrict__ deg)
{
    int e = blockIdx.x * 256 + threadIdx.x;
    if (e >= ETOT) return;
    int t = (e < NEDGES) ? ei[NEDGES + e] : (e - NEDGES);
    atomicAdd(&deg[t], 1);
}

__global__ __launch_bounds__(256)
void scan_block_sum(const int* __restrict__ deg, int* __restrict__ bsum)
{
    __shared__ int sh[256];
    int t = threadIdx.x;
    sh[t] = deg[blockIdx.x * 256 + t];
    __syncthreads();
#pragma unroll
    for (int off = 128; off; off >>= 1) {
        if (t < off) sh[t] += sh[t + off];
        __syncthreads();
    }
    if (t == 0) bsum[blockIdx.x] = sh[0];
}

__global__ __launch_bounds__(256)
void scan_top(const int* __restrict__ bsum, int* __restrict__ boff)
{
    __shared__ int sh[256];
    int t = threadIdx.x;
    int v = bsum[t];
    sh[t] = v;
    __syncthreads();
    for (int off = 1; off < 256; off <<= 1) {
        int u = (t >= off) ? sh[t - off] : 0;
        __syncthreads();
        sh[t] += u;
        __syncthreads();
    }
    boff[t] = sh[t] - v;   // exclusive
}

__global__ __launch_bounds__(256)
void scan_final(const int* __restrict__ deg, const int* __restrict__ boff,
                int* __restrict__ rowptr, int* __restrict__ cursor)
{
    __shared__ int sh[256];
    int t = threadIdx.x;
    int i = blockIdx.x * 256 + t;
    int v = deg[i];
    sh[t] = v;
    __syncthreads();
    for (int off = 1; off < 256; off <<= 1) {
        int u = (t >= off) ? sh[t - off] : 0;
        __syncthreads();
        sh[t] += u;
        __syncthreads();
    }
    int ex = boff[blockIdx.x] + sh[t] - v;
    rowptr[i] = ex;
    cursor[i] = ex;
    if (i == NNODES - 1) rowptr[NNODES] = ETOT;
}

__global__ __launch_bounds__(256)
void csr_fill(const int* __restrict__ ei, int* __restrict__ cursor,
              int* __restrict__ csr_src)
{
    int e = blockIdx.x * 256 + threadIdx.x;
    if (e >= ETOT) return;
    int s = (e < NEDGES) ? ei[e]          : (e - NEDGES);
    int t = (e < NEDGES) ? ei[NEDGES + e] : (e - NEDGES);
    csr_src[atomicAdd(&cursor[t], 1)] = s;
}

// ---------------------------------------------------------------------------
// GAT edge softmax + aggregation, atomic-free gather: block per dst node
// agg[t,:] = bias + sum_p softmax_p(lrelu(hs[src_p]+hd[t])) * h[src_p,:]
// ---------------------------------------------------------------------------
__global__ __launch_bounds__(256)
void gat_gather(const int* __restrict__ rowptr, const int* __restrict__ csr_src,
                const float* __restrict__ hs, const float* __restrict__ hd,
                const float* __restrict__ h, const float* __restrict__ bias,
                float* __restrict__ agg)
{
    __shared__ float salpha[64];
    __shared__ float sm[2];
    int t   = blockIdx.x;
    int tid = threadIdx.x;
    int beg = rowptr[t], end = rowptr[t + 1];
    float hdt = hd[t];

    if (tid == 0) {                  // degree is tiny (~2): serial softmax stats
        float m = -3.4e38f;
        for (int p = beg; p < end; ++p) m = fmaxf(m, lrelu(hs[csr_src[p]] + hdt));
        float ssum = 0.f;
        for (int p = beg; p < end; ++p) ssum += expf(lrelu(hs[csr_src[p]] + hdt) - m);
        sm[0] = m; sm[1] = ssum;
    }
    __syncthreads();
    float m = sm[0], inv = 1.0f / (sm[1] + 1e-16f);

    float acc[4];
#pragma unroll
    for (int i = 0; i < 4; ++i) acc[i] = bias[tid + i * 256];

    for (int base = beg; base < end; base += 64) {
        int cnt = min(64, end - base);
        if (tid < cnt) {
            int s = csr_src[base + tid];
            salpha[tid] = expf(lrelu(hs[s] + hdt) - m) * inv;
        }
        __syncthreads();
        for (int j = 0; j < cnt; ++j) {
            const float* hr = h + (size_t)csr_src[base + j] * D_HID;
            float a = salpha[j];
#pragma unroll
            for (int i = 0; i < 4; ++i) acc[i] += a * hr[tid + i * 256];
        }
        __syncthreads();
    }
#pragma unroll
    for (int i = 0; i < 4; ++i) agg[(size_t)t * D_HID + tid + i * 256] = acc[i];
}

// ---------------------------------------------------------------------------
// Per-row dot products (wave per row): o1 = f(x.w1 + b1), o2 = x.w2
// ---------------------------------------------------------------------------
__global__ __launch_bounds__(256)
void row_dot(const float* __restrict__ x, const float* __restrict__ w1,
             const float* __restrict__ w2, const float* __restrict__ b1,
             float* __restrict__ o1, float* __restrict__ o2, int rows, int sig)
{
    int wave = threadIdx.x >> 5, lane = threadIdx.x & 31;
    int row = blockIdx.x * 8 + wave;
    if (row >= rows) return;
    const float* xr = x + (size_t)row * D_HID;
    float s1 = 0.f, s2 = 0.f;
    for (int d = lane; d < D_HID; d += 32) {
        float v = xr[d];
        s1 += v * w1[d];
        if (w2) s2 += v * w2[d];
    }
#pragma unroll
    for (int msk = 16; msk; msk >>= 1) {
        s1 += __shfl_xor(s1, msk, 32);
        s2 += __shfl_xor(s2, msk, 32);
    }
    if (lane == 0) {
        float r1 = s1 + (b1 ? b1[0] : 0.0f);
        if (sig) r1 = 1.0f / (1.0f + expf(-r1));
        o1[row] = r1;
        if (w2 && o2) o2[row] = s2;
    }
}

// ---------------------------------------------------------------------------
// LayerNorm + ReLU (block per row, D=1024); writes f32 and bf16 copies
// ---------------------------------------------------------------------------
__global__ __launch_bounds__(256)
void ln_relu(const float* __restrict__ src, const float* __restrict__ g,
             const float* __restrict__ be, float* __restrict__ dstf,
             uint16_t* __restrict__ dstb)
{
    __shared__ float rs[8], rs2[8], stat[2];
    int row = blockIdx.x, tid = threadIdx.x, lane = tid & 31, wave = tid >> 5;
    const float* xr = src + (size_t)row * D_HID;
    float v[4], s = 0.f, s2 = 0.f;
#pragma unroll
    for (int i = 0; i < 4; ++i) {
        v[i] = xr[tid + 256 * i];
        s += v[i]; s2 += v[i] * v[i];
    }
#pragma unroll
    for (int m = 16; m; m >>= 1) { s += __shfl_xor(s, m, 32); s2 += __shfl_xor(s2, m, 32); }
    if (lane == 0) { rs[wave] = s; rs2[wave] = s2; }
    __syncthreads();
    if (tid == 0) {
        float a = 0.f, b = 0.f;
#pragma unroll
        for (int i = 0; i < 8; ++i) { a += rs[i]; b += rs2[i]; }
        float mu  = a * (1.0f / D_HID);
        float var = b * (1.0f / D_HID) - mu * mu;
        stat[0] = mu; stat[1] = rsqrtf(var + 1e-5f);
    }
    __syncthreads();
    float mu = stat[0], rsg = stat[1];
#pragma unroll
    for (int i = 0; i < 4; ++i) {
        int d = tid + 256 * i;
        float y = (v[i] - mu) * rsg * g[d] + be[d];
        y = y > 0.f ? y : 0.f;
        dstf[(size_t)row * D_HID + d] = y;
        dstb[(size_t)row * D_HID + d] = f32_to_bf16_bits(y);
    }
}

// ---------------------------------------------------------------------------
// Graph segment bounds (batch is sorted) + atomic-free attention pooling
// ---------------------------------------------------------------------------
__global__ __launch_bounds__(256)
void batch_bounds(const int* __restrict__ batch, int* __restrict__ bp)
{
    int n = blockIdx.x * 256 + threadIdx.x;
    if (n >= NNODES) return;
    int b = batch[n];
    int prev = (n == 0) ? -1 : batch[n - 1];
    for (int q = prev + 1; q <= b; ++q) bp[q] = n;
    if (n == NNODES - 1)
        for (int q = b + 1; q <= NBATCH; ++q) bp[q] = NNODES;
}

__global__ __launch_bounds__(256)
void pool_gather(const float* __restrict__ x, const float* __restrict__ gate,
                 const int* __restrict__ bp, float* __restrict__ p)
{
    __shared__ float red[8], bc[1], lw[256];
    int b = blockIdx.x, tid = threadIdx.x, lane = tid & 31, wave = tid >> 5;
    int beg = bp[b], end = bp[b + 1];

    // segment max of gate
    float m = -3.4e38f;
    for (int n = beg + tid; n < end; n += 256) m = fmaxf(m, gate[n]);
#pragma unroll
    for (int msk = 16; msk; msk >>= 1) m = fmaxf(m, __shfl_xor(m, msk, 32));
    if (lane == 0) red[wave] = m;
    __syncthreads();
    if (tid == 0) {
        float r = red[0];
#pragma unroll
        for (int i = 1; i < 8; ++i) r = fmaxf(r, red[i]);
        bc[0] = r;
    }
    __syncthreads();
    m = bc[0];

    // segment sum of exp
    float s = 0.f;
    for (int n = beg + tid; n < end; n += 256) s += expf(gate[n] - m);
#pragma unroll
    for (int msk = 16; msk; msk >>= 1) s += __shfl_xor(s, msk, 32);
    if (lane == 0) red[wave] = s;
    __syncthreads();
    if (tid == 0) {
        float r = 0.f;
#pragma unroll
        for (int i = 0; i < 8; ++i) r += red[i];
        bc[0] = r;
    }
    __syncthreads();
    float inv = 1.0f / (bc[0] + 1e-16f);

    // weighted sum of node features
    float acc[4] = {0.f, 0.f, 0.f, 0.f};
    for (int base = beg; base < end; base += 256) {
        int cnt = min(256, end - base);
        __syncthreads();
        if (tid < cnt) lw[tid] = expf(gate[base + tid] - m) * inv;
        __syncthreads();
        for (int j = 0; j < cnt; ++j) {
            const float* xr = x + (size_t)(base + j) * D_HID;
            float w = lw[j];
#pragma unroll
            for (int i = 0; i < 4; ++i) acc[i] += w * xr[tid + i * 256];
        }
    }
#pragma unroll
    for (int i = 0; i < 4; ++i) p[(size_t)b * D_HID + tid + i * 256] = acc[i];
}

// ---------------------------------------------------------------------------
extern "C" void kernel_launch(void* const* d_in, const int* in_sizes, int n_in,
                              void* d_out, int out_size, void* d_ws, size_t ws_size,
                              hipStream_t stream)
{
    (void)in_sizes; (void)n_in; (void)out_size; (void)ws_size;
    const float* tree_x = (const float*)d_in[0];
    const int*   ei     = (const int*)d_in[1];
    const int*   batch  = (const int*)d_in[2];
    auto gatW  = [&](int l) { return (const float*)d_in[3 + 4 * l]; };
    auto gatAs = [&](int l) { return (const float*)d_in[4 + 4 * l]; };
    auto gatAd = [&](int l) { return (const float*)d_in[5 + 4 * l]; };
    auto gatB  = [&](int l) { return (const float*)d_in[6 + 4 * l]; };
    auto lnG   = [&](int l) { return (const float*)d_in[23 + 2 * l]; };
    auto lnB   = [&](int l) { return (const float*)d_in[24 + 2 * l]; };
    const float* gate_W = (const float*)d_in[33];
    const float* gate_b = (const float*)d_in[34];
    auto fcnW  = [&](int l) { return (const float*)d_in[35 + 2 * l]; };
    auto fcnB  = [&](int l) { return (const float*)d_in[36 + 2 * l]; };
    auto flnG  = [&](int l) { return (const float*)d_in[45 + 2 * l]; };
    auto flnB  = [&](int l) { return (const float*)d_in[46 + 2 * l]; };
    const float* out_W  = (const float*)d_in[55];
    const float* out_b  = (const float*)d_in[56];
    float* out = (float*)d_out;  // [64]

    // ---- carve workspace ----
    char* ws = (char*)d_ws;
    size_t off = 0;
    auto alloc = [&](size_t bytes) -> void* {
        void* q = ws + off;
        off += (bytes + 255) & ~(size_t)255;
        return q;
    };
    uint16_t* xb      = (uint16_t*)alloc((size_t)NNODES * D_HID * 2);
    float*    h       = (float*)   alloc((size_t)NNODES * D_HID * 4);
    float*    agg     = (float*)   alloc((size_t)NNODES * D_HID * 4);
    uint16_t* wt      = (uint16_t*)alloc((size_t)D_HID * D_HID * 2);
    float*    hs      = (float*)   alloc((size_t)NNODES * 4);
    float*    hd      = (float*)   alloc((size_t)NNODES * 4);
    int*      deg     = (int*)     alloc((size_t)NNODES * 4);
    int*      rowptr  = (int*)     alloc(((size_t)NNODES + 1) * 4);
    int*      cursor  = (int*)     alloc((size_t)NNODES * 4);
    int*      csr_src = (int*)     alloc((size_t)ETOT * 4);
    int*      bsum    = (int*)     alloc(256 * 4);
    int*      boff    = (int*)     alloc(256 * 4);
    int*      bp      = (int*)     alloc((NBATCH + 1) * 4);
    float*    gate    = (float*)   alloc((size_t)NNODES * 4);
    float*    p       = (float*)   alloc((size_t)NBATCH * D_HID * 4);
    uint16_t* pb      = (uint16_t*)alloc((size_t)NBATCH * D_HID * 2);
    float*    hp      = (float*)   alloc((size_t)NBATCH * D_HID * 4);

    dim3 blk(256);

    // ===== one-time CSR over dst (topology is layer-invariant) =====
    fill_i32<<<NNODES / 256, blk, 0, stream>>>(deg, 0, NNODES);
    deg_count<<<ETOT / 256, blk, 0, stream>>>(ei, deg);
    scan_block_sum<<<256, blk, 0, stream>>>(deg, bsum);
    scan_top<<<1, blk, 0, stream>>>(bsum, boff);
    scan_final<<<256, blk, 0, stream>>>(deg, boff, rowptr, cursor);
    csr_fill<<<ETOT / 256, blk, 0, stream>>>(ei, cursor, csr_src);
    batch_bounds<<<NNODES / 256, blk, 0, stream>>>(batch, bp);

    // ===== 5 GAT layers =====
    for (int l = 0; l < 5; ++l) {
        const int K    = (l == 0) ? 17 : D_HID;
        const int Kpad = (l == 0) ? BK : D_HID;   // pad layer-0 K to one K-tile

        wt_convert<<<(D_HID * Kpad + 255) / 256, blk, 0, stream>>>(gatW(l), wt, K, Kpad);
        if (l == 0)
            convert_pad_bf16<<<(unsigned)(((long long)NNODES * Kpad + 255) / 256),
                               blk, 0, stream>>>(tree_x, xb, NNODES, K, Kpad);

        gemm_bf16_wmma<true><<<dim3(NNODES / BM, D_HID / BN), blk, 0, stream>>>(
            xb, wt, nullptr, h, NNODES, Kpad);

        row_dot<<<NNODES / 8, blk, 0, stream>>>(h, gatAs(l), gatAd(l), nullptr,
                                                hs, hd, NNODES, 0);

        gat_gather<<<NNODES, blk, 0, stream>>>(rowptr, csr_src, hs, hd, h, gatB(l), agg);

        ln_relu<<<NNODES, blk, 0, stream>>>(agg, lnG(l), lnB(l), agg, xb);
    }

    // ===== attention pooling =====
    row_dot<<<NNODES / 8, blk, 0, stream>>>(agg, gate_W, nullptr, gate_b,
                                            gate, nullptr, NNODES, 1);
    pool_gather<<<NBATCH, blk, 0, stream>>>(agg, gate, bp, p);

    // ===== 5 FCN layers =====
    convert_pad_bf16<<<(NBATCH * D_HID + 255) / 256, blk, 0, stream>>>(
        p, pb, NBATCH, D_HID, D_HID);
    for (int l = 0; l < 5; ++l) {
        wt_convert<<<(D_HID * D_HID + 255) / 256, blk, 0, stream>>>(fcnW(l), wt,
                                                                    D_HID, D_HID);
        gemm_bf16_wmma<false><<<dim3(1, D_HID / BN), blk, 0, stream>>>(
            pb, wt, fcnB(l), hp, NBATCH, D_HID);
        ln_relu<<<NBATCH, blk, 0, stream>>>(hp, flnG(l), flnB(l), p, pb);
    }

    row_dot<<<NBATCH / 8, blk, 0, stream>>>(p, out_W, nullptr, out_b,
                                            out, nullptr, NBATCH, 0);
}
